// KrauseWindowAttention_49143015801138
// MI455X (gfx1250) — matrix operand order repaired and checked by reading, not compiled, not run on hardware
//
#include <hip/hip_runtime.h>

// ---------------- problem constants ----------------
#define WIN    7
#define SHIFT  3
#define HEADS  4
#define DIM    128
#define HD     32      // head dim
#define NTOK   49      // tokens per window
#define NPAD   64      // padded tokens (4 x 16 tiles)
#define HW     112
#define NWSIDE 16      // 112/7
#define BATCH  16
#define NWTOT  (BATCH * NWSIDE * NWSIDE)   // 4096 windows
#define QKVN   (3 * DIM)                   // 384

typedef __attribute__((ext_vector_type(16))) _Float16 v16h;
typedef __attribute__((ext_vector_type(8)))  _Float16 v8h;
typedef __attribute__((ext_vector_type(4)))  _Float16 v4h;
typedef __attribute__((ext_vector_type(8)))  float    v8f;
typedef __attribute__((ext_vector_type(4)))  unsigned int v4u;
typedef __attribute__((ext_vector_type(8)))  int      v8i;
typedef __attribute__((ext_vector_type(4)))  int      v4i;

// ---------------- Tensor Data Mover: 1-D contiguous tile load to LDS ----------------
// Descriptor per CDNA5 ISA (D# group0 + group1).
// One row of `nelem` 2-byte elements; issued once by a single wave (EXEC-independent).
// Toolchain is the 6-arg (clang-23 / therock) variant:
//   (uint32x4 g0, int32x8 g1, int32x4 g2, int32x4 g3, int32x8 pad, i32 cpol)
#if __has_builtin(__builtin_amdgcn_tensor_load_to_lds)
#define HAVE_TDM 1
__device__ __forceinline__ void tdm_load_1d(const void* gsrc, void* ldst, unsigned nelem) {
  unsigned lds = (unsigned)(unsigned long long)(__SIZE_TYPE__)ldst;  // LDS byte offset
  unsigned long long ga = (unsigned long long)(__SIZE_TYPE__)gsrc;
  v4u g0;
  g0[0] = 1u;                                          // count=1, user descriptor
  g0[1] = lds;                                         // lds_addr [63:32]
  g0[2] = (unsigned)(ga & 0xFFFFFFFFu);                // global_addr[31:0]
  g0[3] = (unsigned)((ga >> 32) & 0x1FFFFFFu) | (2u << 30); // global_addr[56:32] | type=2
  v8i g1;
  g1[0] = (int)(1u << 16);                             // workgroup_mask=0, data_size=1 (2B)
  g1[1] = (int)((nelem & 0xFFFFu) << 16);              // tensor_dim0[15:0] @ bits 63:48
  g1[2] = (int)((nelem >> 16) | (1u << 16));           // tensor_dim0[31:16], tensor_dim1=1
  g1[3] = (int)((nelem & 0xFFFFu) << 16);              // tile_dim0 @ bits 127:112
  g1[4] = 1;                                           // tile_dim1=1, tile_dim2=0
  g1[5] = (int)nelem;                                  // tensor_dim0_stride[31:0]
  g1[6] = 0;
  g1[7] = 0;
  v4i z4 = {0, 0, 0, 0};
  v8i z8 = {0, 0, 0, 0, 0, 0, 0, 0};
  __builtin_amdgcn_tensor_load_to_lds(g0, g1, z4, z4, z8, 0);
}
__device__ __forceinline__ void tdm_drain() {
  __builtin_amdgcn_s_wait_tensorcnt(0);
}
#else
#define HAVE_TDM 0
#endif

// fallback / generic contiguous copy (float4 chunks)
__device__ __forceinline__ void copy16(const void* s, void* d, int n16, int tid, int nthr) {
  const float4* sp = (const float4*)s;
  float4* dp = (float4*)d;
  for (int i = tid; i < n16; i += nthr) dp[i] = sp[i];
}

// ---------------- WMMA fragment helpers (gfx1250 wave32 layouts) ----------------
// A-matrix 16x32 f16 (ISA 7.12.2): lane = {M = lane&15, half = lane>>4}
//   elems 0..7  : K =  8*half .. +7 ; elems 8..15 : K = 16+8*half .. +7
__device__ __forceinline__ v16h frag_a(const _Float16* __restrict__ p, int ld, int lane) {
  const _Float16* r = p + (lane & 15) * ld + ((lane >> 4) << 3);
  v8h lo = *(const v8h*)r;
  v8h hi = *(const v8h*)(r + 16);
  return __builtin_shufflevector(lo, hi, 0,1,2,3,4,5,6,7,8,9,10,11,12,13,14,15);
}
// B-matrix 32x16 f16 from B^T stored row-major [N][K]:
// lane = {N = lane&15, half = lane>>4}; elems 0..7 : K = 8*half .. +7 ; elems 8..15 : K = 16+8*half .. +7
__device__ __forceinline__ v16h frag_b(const _Float16* __restrict__ p, int ld, int lane) {
  const _Float16* r = p + (lane & 15) * ld + ((lane >> 4) << 3);
  v8h lo = *(const v8h*)r;
  v8h hi = *(const v8h*)(r + 16);
  return __builtin_shufflevector(lo, hi, 0,1,2,3,4,5,6,7,8,9,10,11,12,13,14,15);
}
__device__ __forceinline__ v8f wmma16(v16h a, v16h b, v8f c) {
  return __builtin_amdgcn_wmma_f32_16x16x32_f16(false, a, false, b, (short)0, c, false, false);
}

// ---------------- K0: transpose weights to f16 B^T layout ----------------
__global__ __launch_bounds__(256) void k0_weights(const float* __restrict__ wqkv,
                                                  const float* __restrict__ wproj,
                                                  _Float16* __restrict__ wqkvT,
                                                  _Float16* __restrict__ wprojT) {
  int i = blockIdx.x * blockDim.x + threadIdx.x;
  if (i < QKVN * DIM) {                      // wqkvT[n][k] = wqkv[k][n]
    int n = i / DIM, k = i % DIM;
    wqkvT[i] = (_Float16)wqkv[k * QKVN + n];
  }
  int j = i - QKVN * DIM;
  if (j >= 0 && j < DIM * DIM) {             // wprojT[n][k] = wproj[k][n]
    int n = j / DIM, k = j % DIM;
    wprojT[j] = (_Float16)wproj[k * DIM + n];
  }
}

// ---------------- K1: fused roll + window gather + QKV GEMM ----------------
__global__ __launch_bounds__(256) void k1_qkv(const float* __restrict__ x,
                                              const _Float16* __restrict__ wqkvT,
                                              const float* __restrict__ bqkv,
                                              _Float16* __restrict__ Qb,
                                              _Float16* __restrict__ Kb,
                                              _Float16* __restrict__ Vt) {
  extern __shared__ char smem[];
  _Float16* aL = (_Float16*)smem;            // [64][128]
  _Float16* wL = aL + NPAD * DIM;            // [384][128]  (W^T)
  const int win = blockIdx.x;
  const int b = win >> 8, wy = (win >> 4) & 15, wx = win & 15;
  const int tid = threadIdx.x;

  // stage W^T (96 KB): single TDM DMA from wave 0, overlapped with x gather below
#if HAVE_TDM
  if (tid < 32) tdm_load_1d(wqkvT, wL, QKVN * DIM);
#else
  copy16(wqkvT, wL, QKVN * DIM / 8, tid, 256);
#endif
  // prefetch next window's first cacheline (global_prefetch_b8)
  if (win + 1 < NWTOT && tid == 0) {
    int nb = (win + 1) >> 8, nwy = ((win + 1) >> 4) & 15, nwx = (win + 1) & 15;
    int r = (nwy * WIN + SHIFT) % HW, c = (nwx * WIN + SHIFT) % HW;
    __builtin_prefetch(x + (((size_t)nb * HW + r) * HW + c) * DIM, 0, 3);
  }
  // stage rolled x window -> f16 (rows >= 49 zeroed)
  for (int idx = tid; idx < NPAD * (DIM / 4); idx += 256) {
    int tok = idx >> 5, part = idx & 31;
    v4h hv;
    if (tok < NTOK) {
      int ti = tok / WIN, tj = tok % WIN;
      int r = (wy * WIN + ti + SHIFT) % HW;
      int c = (wx * WIN + tj + SHIFT) % HW;
      float4 v = *(const float4*)(x + (((size_t)b * HW + r) * HW + c) * DIM + part * 4);
      hv[0] = (_Float16)v.x; hv[1] = (_Float16)v.y;
      hv[2] = (_Float16)v.z; hv[3] = (_Float16)v.w;
    } else {
      hv[0] = hv[1] = hv[2] = hv[3] = (_Float16)0.f;
    }
    *(v4h*)(aL + tok * DIM + part * 4) = hv;
  }
#if HAVE_TDM
  if (tid < 32) tdm_drain();
#endif
  __syncthreads();

  const int wave = __builtin_amdgcn_readfirstlane(tid >> 5);  // force wave-uniform
  const int lane = tid & 31;
  const float scale = 0.17677669529663687f;  // 32^-0.5
  // 64x384 output: 4 M-tiles x 24 N-tiles; wave handles 3 N-tiles
  for (int t = 0; t < 3; ++t) {
    const int n0 = (wave * 3 + t) * 16;      // uniform
    const int s = n0 >> 7;                   // uniform: q/k/v select
    const int h = (n0 >> 5) & 3;             // uniform: head
    const int d = (n0 & 31) + (lane & 15);   // head-dim column
    const float bb = bqkv[n0 + (lane & 15)];
    const size_t wh = (size_t)win * HEADS + h;
    for (int mt = 0; mt < 4; ++mt) {
      const int m0 = mt * 16;
      v8f acc = {};
#pragma unroll
      for (int k0 = 0; k0 < DIM; k0 += 32) {
        v16h a  = frag_a(aL + m0 * DIM + k0, DIM, lane);
        v16h bf = frag_b(wL + n0 * DIM + k0, DIM, lane);
        acc = wmma16(a, bf, acc);
      }
      const int rbase = m0 + ((lane >> 4) << 3);
      if (s == 0) {                          // scalar branch (uniform)
#pragma unroll
        for (int r = 0; r < 8; ++r)
          Qb[(wh * NPAD + rbase + r) * HD + d] = (_Float16)((acc[r] + bb) * scale);
      } else if (s == 1) {
#pragma unroll
        for (int r = 0; r < 8; ++r)
          Kb[(wh * NPAD + rbase + r) * HD + d] = (_Float16)((acc[r] + bb) * scale);
      } else {                               // V stored transposed [32][64]
#pragma unroll
        for (int r = 0; r < 8; ++r)
          Vt[(wh * HD + d) * NPAD + rbase + r] = (_Float16)(acc[r] + bb);
      }
    }
  }
}

// ---------------- K2: RBF attention per (window, head) ----------------
__global__ __launch_bounds__(128) void k2_attn(const _Float16* __restrict__ Qb,
                                               const _Float16* __restrict__ Kb,
                                               const _Float16* __restrict__ Vt,
                                               const float* __restrict__ log_sigma,
                                               const float* __restrict__ bias_table,
                                               _Float16* __restrict__ Y) {
  extern __shared__ char smem[];
  _Float16* qL = (_Float16*)smem;            // [64][32]
  _Float16* kL = qL + NPAD * HD;             // [64][32]
  _Float16* vL = kL + NPAD * HD;             // [32][64] (V^T)
  float*    sL = (float*)(vL + HD * NPAD);   // [64][64] scores f32
  _Float16* pL = (_Float16*)(sL + NPAD * NPAD); // [64][64] probs f16
  float*   qsq = (float*)(pL + NPAD * NPAD); // [64]
  float*   ksq = qsq + NPAD;                 // [64]

  const int wh = blockIdx.x;
  const int win = wh >> 2, h = wh & 3;
  const int wy = (win >> 4) & 15, wx = win & 15;
  const int tid = threadIdx.x;

  // stage q, k, v^T (4 KB each): three TDM DMAs from wave 0
#if HAVE_TDM
  if (tid < 32) {
    tdm_load_1d(Qb + (size_t)wh * NPAD * HD, qL, NPAD * HD);
    tdm_load_1d(Kb + (size_t)wh * NPAD * HD, kL, NPAD * HD);
    tdm_load_1d(Vt + (size_t)wh * HD * NPAD, vL, HD * NPAD);
    tdm_drain();
  }
#else
  copy16(Qb + (size_t)wh * NPAD * HD, qL, NPAD * HD / 8, tid, 128);
  copy16(Kb + (size_t)wh * NPAD * HD, kL, NPAD * HD / 8, tid, 128);
  copy16(Vt + (size_t)wh * HD * NPAD, vL, HD * NPAD / 8, tid, 128);
#endif
  __syncthreads();
  // squared norms of (already scaled) q and k rows
  if (tid < NPAD) {
    float s = 0.f;
    for (int d = 0; d < HD; ++d) { float t = (float)qL[tid * HD + d]; s += t * t; }
    qsq[tid] = s;
  } else {
    int n = tid - NPAD; float s = 0.f;
    for (int d = 0; d < HD; ++d) { float t = (float)kL[n * HD + d]; s += t * t; }
    ksq[n] = s;
  }
  __syncthreads();

  const int wave = __builtin_amdgcn_readfirstlane(tid >> 5);
  const int lane = tid & 31;
  // S = Q K^T : wave owns row-tile `wave`; K-dim = 32 -> single WMMA per tile
  {
    const int m0 = wave * 16;
    v16h a = frag_a(qL + m0 * HD, HD, lane);
#pragma unroll
    for (int nt = 0; nt < 4; ++nt) {
      v16h bf = frag_b(kL + nt * 16 * HD, HD, lane);  // B^T[N=m][K=d] == kL
      v8f acc = {};
      acc = wmma16(a, bf, acc);
      const int cn = nt * 16 + (lane & 15);
      const int rb = m0 + ((lane >> 4) << 3);
#pragma unroll
      for (int r = 0; r < 8; ++r) sL[(rb + r) * NPAD + cn] = acc[r];
    }
  }
  __syncthreads();

  // softmax epilogue: RBF transform + rel-pos bias + shift mask (analytic)
  const float coef = -0.5f * __expf(-2.0f * log_sigma[h]);
  if (tid < NPAD) {
    const int n = tid;
    if (n < NTOK) {
      const int ni = n / WIN, nj = n % WIN;
      const int sr = wy * WIN + ni, sc = wx * WIN + nj;
      const int rg = (sr < HW - WIN) ? 0 : ((sr < HW - SHIFT) ? 1 : 2);
      const int cg = (sc < HW - WIN) ? 0 : ((sc < HW - SHIFT) ? 1 : 2);
      const int gn = rg * 3 + cg;
      float mx = -1e30f;
      for (int m = 0; m < NTOK; ++m) {
        const int mi = m / WIN, mj = m % WIN;
        float dist = qsq[n] + ksq[m] - 2.0f * sL[n * NPAD + m];
        dist = fmaxf(dist, 1e-12f);
        const int rel = (ni - mi + WIN - 1) * (2 * WIN - 1) + (nj - mj + WIN - 1);
        float v = coef * dist + bias_table[rel * HEADS + h];
        const int mr = wy * WIN + mi, mc = wx * WIN + mj;
        const int mrg = (mr < HW - WIN) ? 0 : ((mr < HW - SHIFT) ? 1 : 2);
        const int mcg = (mc < HW - WIN) ? 0 : ((mc < HW - SHIFT) ? 1 : 2);
        if (mrg * 3 + mcg != gn) v -= 100.0f;
        sL[n * NPAD + m] = v;
        mx = fmaxf(mx, v);
      }
      float sum = 0.f;
      for (int m = 0; m < NTOK; ++m) {
        float e = __expf(sL[n * NPAD + m] - mx);
        sL[n * NPAD + m] = e;
        sum += e;
      }
      const float inv = 1.0f / sum;
      for (int m = 0; m < NTOK; ++m) pL[n * NPAD + m] = (_Float16)(sL[n * NPAD + m] * inv);
      for (int m = NTOK; m < NPAD; ++m) pL[n * NPAD + m] = (_Float16)0.f;
    } else {
      for (int m = 0; m < NPAD; ++m) pL[n * NPAD + m] = (_Float16)0.f;
    }
  }
  __syncthreads();

  // out = P · V : wave owns row-tile `wave`; 2 d-tiles, K = 64 -> 2 WMMA steps
  {
    const int m0 = wave * 16;
#pragma unroll
    for (int dt = 0; dt < 2; ++dt) {
      v8f acc = {};
#pragma unroll
      for (int k0 = 0; k0 < NPAD; k0 += 32) {
        v16h a  = frag_a(pL + m0 * NPAD + k0, NPAD, lane);
        v16h bf = frag_b(vL + dt * 16 * NPAD + k0, NPAD, lane);  // B^T[d][m] == vL
        acc = wmma16(a, bf, acc);
      }
      const int dcol = dt * 16 + (lane & 15);
      const int rb = m0 + ((lane >> 4) << 3);
#pragma unroll
      for (int r = 0; r < 8; ++r) {
        const int row = rb + r;
        Y[((size_t)win * NPAD + row) * DIM + h * HD + dcol] = (_Float16)acc[r];
      }
    }
  }
}

// ---------------- K3: output projection + inverse roll ----------------
__global__ __launch_bounds__(256) void k3_proj(const _Float16* __restrict__ Y,
                                               const _Float16* __restrict__ wprojT,
                                               const float* __restrict__ bproj,
                                               float* __restrict__ out) {
  extern __shared__ char smem[];
  _Float16* yL = (_Float16*)smem;            // [64][128]
  _Float16* wL = yL + NPAD * DIM;            // [128][128] (W^T)
  const int win = blockIdx.x;
  const int b = win >> 8, wy = (win >> 4) & 15, wx = win & 15;
  const int tid = threadIdx.x;

#if HAVE_TDM
  if (tid < 32) {
    tdm_load_1d(Y + (size_t)win * NPAD * DIM, yL, NPAD * DIM);
    tdm_load_1d(wprojT, wL, DIM * DIM);
    tdm_drain();
  }
#else
  copy16(Y + (size_t)win * NPAD * DIM, yL, NPAD * DIM / 8, tid, 256);
  copy16(wprojT, wL, DIM * DIM / 8, tid, 256);
#endif
  __syncthreads();

  const int wave = __builtin_amdgcn_readfirstlane(tid >> 5);
  const int lane = tid & 31;
  const int n0 = wave * 16;                  // wave owns one N-tile column
  const int c = n0 + (lane & 15);
  const float bb = bproj[c];
  for (int mt = 0; mt < 4; ++mt) {
    const int m0 = mt * 16;
    v8f acc = {};
#pragma unroll
    for (int k0 = 0; k0 < DIM; k0 += 32) {
      v16h a  = frag_a(yL + m0 * DIM + k0, DIM, lane);
      v16h bf = frag_b(wL + n0 * DIM + k0, DIM, lane);
      acc = wmma16(a, bf, acc);
    }
    const int rb = m0 + ((lane >> 4) << 3);
#pragma unroll
    for (int r = 0; r < 8; ++r) {
      const int row = rb + r;
      if (row < NTOK) {
        const int ti = row / WIN, tj = row % WIN;
        const int orr = (wy * WIN + ti + SHIFT) % HW;
        const int oc  = (wx * WIN + tj + SHIFT) % HW;
        out[(((size_t)b * HW + orr) * HW + oc) * DIM + c] = acc[r] + bb;
      }
    }
  }
}

// ---------------- launcher ----------------
extern "C" void kernel_launch(void* const* d_in, const int* in_sizes, int n_in,
                              void* d_out, int out_size, void* d_ws, size_t ws_size,
                              hipStream_t stream) {
  const float* x         = (const float*)d_in[0];
  const float* w_qkv     = (const float*)d_in[1];
  const float* b_qkv     = (const float*)d_in[2];
  const float* w_proj    = (const float*)d_in[3];
  const float* b_proj    = (const float*)d_in[4];
  const float* log_sigma = (const float*)d_in[5];
  const float* bias_tab  = (const float*)d_in[6];
  float* out = (float*)d_out;

  char* ws = (char*)d_ws;
  size_t off = 0;
  _Float16* wqkvT  = (_Float16*)(ws + off); off += (size_t)QKVN * DIM * 2;
  _Float16* wprojT = (_Float16*)(ws + off); off += (size_t)DIM * DIM * 2;
  off = (off + 255) & ~(size_t)255;
  _Float16* Qb = (_Float16*)(ws + off); off += (size_t)NWTOT * HEADS * NPAD * HD * 2;
  _Float16* Kb = (_Float16*)(ws + off); off += (size_t)NWTOT * HEADS * NPAD * HD * 2;
  _Float16* Vt = (_Float16*)(ws + off); off += (size_t)NWTOT * HEADS * HD * NPAD * 2;
  _Float16* Yb = (_Float16*)(ws + off); off += (size_t)NWTOT * NPAD * DIM * 2;

  {
    int total = QKVN * DIM + DIM * DIM;
    k0_weights<<<(total + 255) / 256, 256, 0, stream>>>(w_qkv, w_proj, wqkvT, wprojT);
  }
  {
    size_t shm = (size_t)(NPAD * DIM + QKVN * DIM) * sizeof(_Float16);
    k1_qkv<<<NWTOT, 256, shm, stream>>>(x, wqkvT, b_qkv, Qb, Kb, Vt);
  }
  {
    size_t shm = (size_t)(NPAD * HD * 2 + HD * NPAD) * sizeof(_Float16)
               + (size_t)NPAD * NPAD * sizeof(float)
               + (size_t)NPAD * NPAD * sizeof(_Float16)
               + 2 * NPAD * sizeof(float);
    k2_attn<<<NWTOT * HEADS, 128, shm, stream>>>(Qb, Kb, Vt, log_sigma, bias_tab, Yb);
  }
  {
    size_t shm = (size_t)(NPAD * DIM + DIM * DIM) * sizeof(_Float16);
    k3_proj<<<NWTOT, 256, shm, stream>>>(Yb, wprojT, b_proj, out);
  }
}
